// Net_88321707475068
// MI455X (gfx1250) — matrix operand order repaired and checked by reading, not compiled, not run on hardware
//
#include <hip/hip_runtime.h>
#include <math.h>

typedef __attribute__((ext_vector_type(16))) _Float16 v16h;
typedef __attribute__((ext_vector_type(8)))  _Float16 v8h;
typedef __attribute__((ext_vector_type(8)))  float    v8f;

static constexpr int kB = 128;   // batches (graphs)
static constexpr int kN = 512;   // nodes per graph
static constexpr int kH = 64;    // hidden features
static constexpr int kK = 16;    // kNN neighbors
static constexpr int kF = 5;     // input features
static constexpr int kC = 10;    // classes
static constexpr int kBN = kB * kN;

__device__ __forceinline__ float eluf(float v) { return v > 0.0f ? v : expm1f(v); }

// D(f32 16x16) = A(f16 16x32) * B(f16 32x16) + C ; wave32, f32 accumulate
__device__ __forceinline__ v8f wmma16(v16h a, v16h b, v8f c) {
  return __builtin_amdgcn_wmma_f32_16x16x32_f16(false, a, false, b, (short)0, c,
                                                false, false);
}

// Build an A-matrix (16x32 f16) register tile from a row pointer.
// wave32 layout: lane<16 (row=lane)   holds K {koff+0..15}
//                lane>=16 (row=lane-16) holds K {koff+16..31}
__device__ __forceinline__ v16h load_a_tile(const _Float16* __restrict__ row,
                                            int half, int koff) {
  const v8h lo = *(const v8h*)(row + koff + half * 16);
  const v8h hi = *(const v8h*)(row + koff + 8 + half * 16);
  v16h a;
  #pragma unroll
  for (int t = 0; t < 8; ++t) { a[t] = lo[t]; a[8 + t] = hi[t]; }
  return a;
}

// ---------------------------------------------------------------------------
// Kernel 1: input MLP  (x[BN,5] -> h[BN,64]), feature-parallel, LDS staged
// block = 64 threads, each block processes 8 nodes
// ---------------------------------------------------------------------------
__global__ void in_mlp_kernel(const float* __restrict__ x,
                              const float* __restrict__ W0, const float* __restrict__ b0,
                              const float* __restrict__ W1, const float* __restrict__ b1,
                              const float* __restrict__ W2, const float* __restrict__ b2,
                              float* __restrict__ hout)
{
  __shared__ float w0s[kF * kH], w1s[kH * kH], w2s[kH * kH];
  __shared__ float b0s[kH], b1s[kH], b2s[kH];
  __shared__ float xs[kF], t1[kH], t2[kH];
  const int tid = threadIdx.x;  // 64
  for (int t = tid; t < kF * kH; t += kH) w0s[t] = W0[t];
  for (int t = tid; t < kH * kH; t += kH) { w1s[t] = W1[t]; w2s[t] = W2[t]; }
  b0s[tid] = b0[tid]; b1s[tid] = b1[tid]; b2s[tid] = b2[tid];
  __syncthreads();
  for (int it = 0; it < 8; ++it) {
    const size_t node = (size_t)blockIdx.x * 8 + it;
    if (tid < kF) xs[tid] = x[node * kF + tid];
    __syncthreads();
    float a = b0s[tid];
    #pragma unroll
    for (int f = 0; f < kF; ++f) a += xs[f] * w0s[f * kH + tid];
    t1[tid] = eluf(a);
    __syncthreads();
    a = b1s[tid];
    for (int f = 0; f < kH; ++f) a += t1[f] * w1s[f * kH + tid];
    t2[tid] = eluf(a);
    __syncthreads();
    a = b2s[tid];
    for (int f = 0; f < kH; ++f) a += t2[f] * w2s[f * kH + tid];
    hout[node * kH + tid] = eluf(a);
    __syncthreads();
  }
}

// ---------------------------------------------------------------------------
// Kernel 2: prep — split-f16 copy of h (hi + residual lo) + exact f32 norms.
// block=256 (8 waves = 8 nodes)
// ---------------------------------------------------------------------------
__global__ void prep_kernel(const float* __restrict__ h,
                            _Float16* __restrict__ hhi, _Float16* __restrict__ hlo,
                            float* __restrict__ sq)
{
  const int lane = threadIdx.x & 31;
  const size_t node = (size_t)blockIdx.x * 8 + (threadIdx.x >> 5);
  const float* hp = h + node * kH;
  const float v0 = hp[lane];
  const float v1 = hp[lane + 32];
  const _Float16 h0 = (_Float16)v0;
  const _Float16 h1 = (_Float16)v1;
  hhi[node * kH + lane]      = h0;
  hhi[node * kH + lane + 32] = h1;
  hlo[node * kH + lane]      = (_Float16)(v0 - (float)h0);
  hlo[node * kH + lane + 32] = (_Float16)(v1 - (float)h1);
  float s = v0 * v0 + v1 * v1;
  #pragma unroll
  for (int off = 16; off > 0; off >>= 1) s += __shfl_xor(s, off, 32);
  if (lane == 0) sq[node] = s;
}

// ---------------------------------------------------------------------------
// Kernel 3: kNN — one wave per (batch, 16-row tile).
// Compensated-f16 Gram via v_wmma_f32_16x16x32_f16:
//   dot(hi+lo, hi+lo) ~= hi*hi + lo*hi + hi*lo   (lo*lo ~ 2^-22, dropped)
// -> near-f32 distance ranking (reference ranks exact f32 distances, and the
//    gap between adjacent neighbor order-statistics (~1e-2) is >> 1e-6 error).
// scores d = |i|^2 + |j|^2 - 2*dot into LDS, diagonal masked (ref adds 1e30),
// then 16x argmin with cross-lane shfl reduction (index tie-break = smaller).
// WMMA wave32 layouts:
//   A 16x32: lane<16 row=lane holds K{0..15}; lane>=16 row=lane-16 K{16..31} (same order as B)
//   B 32x16: lane holds column lane%16; K = half*16 .. +15 contiguous (one v16h)
//   C 16x16: lane holds column lane%16; VGPR r -> row r + 8*(lane>=16)
// ---------------------------------------------------------------------------
__global__ void knn_kernel(const _Float16* __restrict__ hhi,
                           const _Float16* __restrict__ hlo,
                           const float* __restrict__ sq,
                           int* __restrict__ nbr)
{
  __shared__ float sd[16][kN];   // 32 KB score rows
  __shared__ float sqs[kN];
  const int lane = threadIdx.x;
  const int b  = blockIdx.x >> 5;           // 32 i-tiles per batch
  const int i0 = (blockIdx.x & 31) * 16;
  const int half = lane >> 4, r = lane & 15;
  const _Float16* hbh = hhi + (size_t)b * kN * kH;
  const _Float16* hbl = hlo + (size_t)b * kN * kH;
  for (int t = lane; t < kN; t += 32) sqs[t] = sq[b * kN + t];

  const _Float16* arowh = hbh + (size_t)(i0 + r) * kH;
  const _Float16* arowl = hbl + (size_t)(i0 + r) * kH;
  const v16h a0h = load_a_tile(arowh, half, 0);
  const v16h a1h = load_a_tile(arowh, half, 32);
  const v16h a0l = load_a_tile(arowl, half, 0);
  const v16h a1l = load_a_tile(arowl, half, 32);
  __syncthreads();

  for (int jt = 0; jt < kN / 16; ++jt) {
    const int j0 = jt * 16;
    const _Float16* browh = hbh + (size_t)(j0 + r) * kH;
    const _Float16* browl = hbl + (size_t)(j0 + r) * kH;
    const v16h bh0 = *(const v16h*)(browh + half * 16);
    const v16h bh1 = *(const v16h*)(browh + 32 + half * 16);
    const v16h bl0 = *(const v16h*)(browl + half * 16);
    const v16h bl1 = *(const v16h*)(browl + 32 + half * 16);
    v8f c = {};
    c = wmma16(a0h, bh0, c);   // hi * hi
    c = wmma16(a1h, bh1, c);
    c = wmma16(a0l, bh0, c);   // lo * hi
    c = wmma16(a1l, bh1, c);
    c = wmma16(a0h, bl0, c);   // hi * lo
    c = wmma16(a1h, bl1, c);
    const float sj = sqs[j0 + r];
    #pragma unroll
    for (int vr = 0; vr < 8; ++vr) {
      const int row = vr + half * 8;
      sd[row][j0 + r] = sqs[i0 + row] + sj - 2.0f * c[vr];
    }
  }
  __syncthreads();
  if (lane < 16) sd[lane][i0 + lane] = 1e30f;  // mask self
  __syncthreads();

  for (int row = 0; row < 16; ++row) {
    int* orow = nbr + ((size_t)b * kN + i0 + row) * kK;
    for (int t = 0; t < kK; ++t) {
      float v = 1e38f; int id = 0x7fffffff;
      for (int c2 = lane; c2 < kN; c2 += 32) {
        const float s = sd[row][c2];
        if (s < v) { v = s; id = c2; }
      }
      #pragma unroll
      for (int off = 16; off > 0; off >>= 1) {
        const float ov = __shfl_xor(v, off, 32);
        const int  oid = __shfl_xor(id, off, 32);
        if (ov < v || (ov == v && oid < id)) { v = ov; id = oid; }
      }
      if (lane == 0) orow[t] = id;
      if (lane == (id & 31)) sd[row][id] = 1e30f;   // remove winner
      __syncthreads();
    }
  }
}

// ---------------------------------------------------------------------------
// Kernel 4: EdgeConv — 4 waves/block, one node per wave (x4 iterations).
// m = [xi, xj - xi] (16x128) built directly in WMMA A layout;
// W0 (128x64) / W1 (64x64) transposed+padded in LDS so every B tile is one
// contiguous 32B v16h read; layer-1 output transposed C->A via padded LDS.
// ---------------------------------------------------------------------------
__global__ void edgeconv_kernel(const float* __restrict__ hin, const int* __restrict__ nbr,
                                const float* __restrict__ W0, const float* __restrict__ Bb0,
                                const float* __restrict__ W1, const float* __restrict__ Bb1,
                                float* __restrict__ hout)
{
  constexpr int WS0 = 144;  // padded k-stride for W0t [n][k<128]
  constexpr int WS1 = 80;   // padded k-stride for W1t [n][k<64]
  constexpr int ES  = 72;   // padded n-stride for e-buffer [m][n<64]
  __shared__ __attribute__((aligned(32))) _Float16 w0t[kH * WS0];
  __shared__ __attribute__((aligned(32))) _Float16 w1t[kH * WS1];
  __shared__ float b0s[kH], b1s[kH];
  __shared__ __attribute__((aligned(32))) _Float16 ebuf[4][16 * ES];

  const int tid = threadIdx.x;  // 128
  for (int t = tid; t < kH * 2 * kH; t += 128) {
    const int n = t >> 7, k = t & 127;
    w0t[n * WS0 + k] = (_Float16)W0[k * kH + n];
  }
  for (int t = tid; t < kH * kH; t += 128) {
    const int n = t >> 6, k = t & 63;
    w1t[n * WS1 + k] = (_Float16)W1[k * kH + n];
  }
  if (tid < kH) { b0s[tid] = Bb0[tid]; b1s[tid] = Bb1[tid]; }
  __syncthreads();

  const int wave = tid >> 5, lane = tid & 31;
  const int half = lane >> 4, r = lane & 15;
  _Float16* eb = &ebuf[wave][0];

  for (int it = 0; it < 4; ++it) {
    const size_t node = (size_t)blockIdx.x * 16 + it * 4 + wave;
    const float* hi = hin + node * kH;
    const int bi = (int)(node >> 9);              // node / kN
    const int j  = nbr[node * kK + r];            // neighbor row for edge r
    const float* hj = hin + ((size_t)bi * kN + j) * kH;

    // ---- layer 1: e1 = elu(m @ W0 + b0), m = [xi, xj-xi] (16x128) ----
    v8f c1[4] = {};
    #pragma unroll
    for (int kt = 0; kt < 4; ++kt) {
      const int o0 = kt * 32 + half * 16, o1 = o0 + 8;
      v16h a;
      if (kt < 2) {                              // columns < 64: xi (broadcast over edges)
        #pragma unroll
        for (int t = 0; t < 8; ++t) {
          a[t]     = (_Float16)hi[o0 + t];
          a[8 + t] = (_Float16)hi[o1 + t];
        }
      } else {                                   // columns >= 64: xj - xi
        const int q0 = o0 - kH, q1 = o1 - kH;
        #pragma unroll
        for (int t = 0; t < 8; ++t) {
          a[t]     = (_Float16)(hj[q0 + t] - hi[q0 + t]);
          a[8 + t] = (_Float16)(hj[q1 + t] - hi[q1 + t]);
        }
      }
      #pragma unroll
      for (int nt = 0; nt < 4; ++nt) {
        const v16h bm = *(const v16h*)&w0t[(nt * 16 + r) * WS0 + kt * 32 + half * 16];
        c1[nt] = wmma16(a, bm, c1[nt]);
      }
    }
    // epilogue: bias + elu, store transposed (C layout -> e[m][n]) for layer-2 A
    #pragma unroll
    for (int nt = 0; nt < 4; ++nt) {
      const int n = nt * 16 + r;
      const float bias = b0s[n];
      #pragma unroll
      for (int vr = 0; vr < 8; ++vr) {
        const int m = vr + half * 8;
        eb[m * ES + n] = (_Float16)eluf(c1[nt][vr] + bias);
      }
    }
    __syncthreads();

    // ---- layer 2: e2 = elu(e1 @ W1 + b1) ----
    v8f c2[4] = {};
    #pragma unroll
    for (int kt = 0; kt < 2; ++kt) {
      const int o0 = kt * 32 + half * 16, o1 = o0 + 8;
      const _Float16* ep = eb + r * ES;
      const v8h lo  = *(const v8h*)(ep + o0);
      const v8h hi2 = *(const v8h*)(ep + o1);
      v16h a;
      #pragma unroll
      for (int t = 0; t < 8; ++t) { a[t] = lo[t]; a[8 + t] = hi2[t]; }
      #pragma unroll
      for (int nt = 0; nt < 4; ++nt) {
        const v16h bm = *(const v16h*)&w1t[(nt * 16 + r) * WS1 + kt * 32 + half * 16];
        c2[nt] = wmma16(a, bm, c2[nt]);
      }
    }
    // epilogue: bias + elu, aggregate sum over 16 edges, write h_next
    #pragma unroll
    for (int nt = 0; nt < 4; ++nt) {
      const int n = nt * 16 + r;
      const float bias = b1s[n];
      float s = 0.0f;
      #pragma unroll
      for (int vr = 0; vr < 8; ++vr) s += eluf(c2[nt][vr] + bias);
      s += __shfl_xor(s, 16, 32);               // combine both row-halves
      if (half == 0) hout[node * kH + n] = s;
    }
    __syncthreads();                            // ebuf reuse next iteration
  }
}

// ---------------------------------------------------------------------------
// Kernel 5: global max pool over nodes
// ---------------------------------------------------------------------------
__global__ void pool_kernel(const float* __restrict__ h, float* __restrict__ g)
{
  const int b = blockIdx.x, tid = threadIdx.x;  // 64
  const float* p = h + (size_t)b * kN * kH + tid;
  float m = -3.4e38f;
  for (int n = 0; n < kN; ++n) m = fmaxf(m, p[(size_t)n * kH]);
  g[b * kH + tid] = m;
}

// ---------------------------------------------------------------------------
// Kernel 6: output MLP + log_softmax (one block per graph)
// ---------------------------------------------------------------------------
__global__ void out_mlp_kernel(const float* __restrict__ g,
                               const float* __restrict__ W0, const float* __restrict__ b0,
                               const float* __restrict__ W1, const float* __restrict__ b1,
                               const float* __restrict__ W2, const float* __restrict__ b2,
                               float* __restrict__ out)
{
  __shared__ float w0s[kH * kH], w1s[kH * kH], w2s[kH * kC];
  __shared__ float b0s[kH], b1s[kH], b2s[kC];
  __shared__ float t1[kH], t2[kH], lg[kC], lsev;
  const int tid = threadIdx.x;   // 64
  const int row = blockIdx.x;
  for (int t = tid; t < kH * kH; t += kH) { w0s[t] = W0[t]; w1s[t] = W1[t]; }
  for (int t = tid; t < kH * kC; t += kH) w2s[t] = W2[t];
  b0s[tid] = b0[tid]; b1s[tid] = b1[tid];
  if (tid < kC) b2s[tid] = b2[tid];
  __syncthreads();
  t1[tid] = g[(size_t)row * kH + tid];
  __syncthreads();
  float a = b0s[tid];
  for (int f = 0; f < kH; ++f) a += t1[f] * w0s[f * kH + tid];
  t2[tid] = eluf(a);
  __syncthreads();
  a = b1s[tid];
  for (int f = 0; f < kH; ++f) a += t2[f] * w1s[f * kH + tid];
  t1[tid] = eluf(a);
  __syncthreads();
  if (tid < kC) {
    float a2 = b2s[tid];
    for (int f = 0; f < kH; ++f) a2 += t1[f] * w2s[f * kC + tid];
    lg[tid] = a2;
  }
  __syncthreads();
  if (tid == 0) {
    float m = lg[0];
    for (int t = 1; t < kC; ++t) m = fmaxf(m, lg[t]);
    float s = 0.0f;
    for (int t = 0; t < kC; ++t) s += expf(lg[t] - m);
    lsev = m + logf(s);
  }
  __syncthreads();
  if (tid < kC) out[(size_t)row * kC + tid] = lg[tid] - lsev;
}

// ---------------------------------------------------------------------------
extern "C" void kernel_launch(void* const* d_in, const int* in_sizes, int n_in,
                              void* d_out, int out_size, void* d_ws, size_t ws_size,
                              hipStream_t stream)
{
  (void)in_sizes; (void)n_in; (void)out_size; (void)ws_size;
  const float* x    = (const float*)d_in[0];
  const float* Win0 = (const float*)d_in[1];  const float* bin0 = (const float*)d_in[2];
  const float* Win1 = (const float*)d_in[3];  const float* bin1 = (const float*)d_in[4];
  const float* Win2 = (const float*)d_in[5];  const float* bin2 = (const float*)d_in[6];
  const float* We0  = (const float*)d_in[7];  const float* be0  = (const float*)d_in[8];
  const float* We1  = (const float*)d_in[9];  const float* be1  = (const float*)d_in[10];
  const float* Wo0  = (const float*)d_in[11]; const float* bo0  = (const float*)d_in[12];
  const float* Wo1  = (const float*)d_in[13]; const float* bo1  = (const float*)d_in[14];
  const float* Wo2  = (const float*)d_in[15]; const float* bo2  = (const float*)d_in[16];

  char* ws = (char*)d_ws;
  float*    hA  = (float*)ws;    ws += (size_t)kBN * kH * sizeof(float);
  float*    hB  = (float*)ws;    ws += (size_t)kBN * kH * sizeof(float);
  _Float16* hhi = (_Float16*)ws; ws += (size_t)kBN * kH * sizeof(_Float16);
  _Float16* hlo = (_Float16*)ws; ws += (size_t)kBN * kH * sizeof(_Float16);
  float*    sq  = (float*)ws;    ws += (size_t)kBN * sizeof(float);
  int*      nbr = (int*)ws;      ws += (size_t)kBN * kK * sizeof(int);
  float*    g   = (float*)ws;    ws += (size_t)kB * kH * sizeof(float);

  in_mlp_kernel<<<kBN / 8, 64, 0, stream>>>(x, Win0, bin0, Win1, bin1, Win2, bin2, hA);

  const float* cur = hA;
  float* nxt = hB;
  for (int l = 0; l < 2; ++l) {
    prep_kernel<<<kBN / 8, 256, 0, stream>>>(cur, hhi, hlo, sq);
    knn_kernel<<<kB * (kN / 16), 32, 0, stream>>>(hhi, hlo, sq, nbr);
    edgeconv_kernel<<<kBN / 16, 128, 0, stream>>>(
        cur, nbr,
        We0 + (size_t)l * 2 * kH * kH, be0 + (size_t)l * kH,
        We1 + (size_t)l * kH * kH,     be1 + (size_t)l * kH, nxt);
    float* tmp = (float*)cur; cur = nxt; nxt = tmp;
  }

  pool_kernel<<<kB, kH, 0, stream>>>(cur, g);
  out_mlp_kernel<<<kB, kH, 0, stream>>>(g, Wo0, bo0, Wo1, bo1, Wo2, bo2, (float*)d_out);
}
